// GLAMemoryCompressor_14998025798088
// MI455X (gfx1250) — matrix operand order repaired and checked by reading, not compiled, not run on hardware
//
#include <hip/hip_runtime.h>
#include <hip/hip_bf16.h>

// Problem dims (fixed by the reference)
#define HB   16
#define TDIM 4096
#define BDIM 4
#define DDIM 2048
#define DK   64
#define DV   128
#define NK   1024   // H*DK
#define NV   2048   // H*DV
#define BT   16384  // B*T

// ---- workspace layout (bytes) ----
#define OFF_X   0ull                         // x bf16         [BT][D]      64 MiB
#define OFF_KW  67108864ull                  // k_w bf16       [NK][D]       4 MiB
#define OFF_VW  71303168ull                  // v_w bf16       [NV][D]       8 MiB
#define OFF_GK  79691776ull                  // gk/decay f32   [BT][NK]     64 MiB
#define OFF_KD  146800640ull                 // k*decay bf16 T [NK][BT]     32 MiB
#define OFF_VT  180355072ull                 // v bf16 T       [NV][BT]     64 MiB
#define OFF_EG  247463936ull                 // exp(g_total)   [B][NK]      16 KiB

typedef __attribute__((ext_vector_type(16))) __bf16 v16bf;
typedef __attribute__((ext_vector_type(8)))  float  v8f;
typedef __attribute__((ext_vector_type(4)))  int    v4i_t;

union Frag  { v16bf v; uint4 q[2]; };
union Pack8 { __bf16 h[8]; uint4 q; };

static __device__ __forceinline__ uint4 ld16(const __bf16* p) {
  return *reinterpret_cast<const uint4*>(p);
}

// ---- CDNA5 async global->LDS copy (ASYNCcnt path) ----
static __device__ __forceinline__ void async_ld16(const void* g, void* l) {
#if __has_builtin(__builtin_amdgcn_global_load_async_to_lds_b128)
  void* gnc = const_cast<void*>(g);
  __builtin_amdgcn_global_load_async_to_lds_b128(
      (__attribute__((address_space(1))) v4i_t*)gnc,
      (__attribute__((address_space(3))) v4i_t*)l, 0, 0);
#else
  unsigned lo = (unsigned)(__UINTPTR_TYPE__)(__attribute__((address_space(3))) void*)l;
  asm volatile("global_load_async_to_lds_b128 %0, %1, off"
               :: "v"(lo), "v"(g) : "memory");
#endif
}

#if __has_builtin(__builtin_amdgcn_s_wait_asynccnt)
#define WAIT_ASYNC(n) __builtin_amdgcn_s_wait_asynccnt(n)
#else
#define WAIT_ASYNC(n) asm volatile("s_wait_asynccnt %0" :: "n"(n) : "memory")
#endif

// ---------------- K0: f32 -> bf16 weight conversion ----------------
__global__ void cvt_f32_bf16(const float* __restrict__ in, __bf16* __restrict__ out, int n) {
  int i = blockIdx.x * blockDim.x + threadIdx.x;
  if (i < n) out[i] = (__bf16)in[i];
}

// ---------------- K1: RMSNorm + gate MLP ----------------
__global__ void __launch_bounds__(256) rmsnorm_gate(
    const float* __restrict__ hs, const float* __restrict__ norm_w,
    const float* __restrict__ gk_w1, const float* __restrict__ gk_w2,
    const float* __restrict__ gk_b,
    __bf16* __restrict__ xq, float* __restrict__ gk)
{
  __shared__ float smem[4096];   // 16 KiB
  const int row = blockIdx.x;
  const int tid = threadIdx.x;
  const float* xr = hs + (size_t)row * DDIM;

  float v[8];
  float ss = 0.f;
#pragma unroll
  for (int i = 0; i < 8; ++i) {
    float t = xr[tid + i * 256];
    v[i] = t; ss += t * t;
  }
  smem[tid] = ss;
  __syncthreads();
  for (int off = 128; off > 0; off >>= 1) {
    if (tid < off) smem[tid] += smem[tid + off];
    __syncthreads();
  }
  const float inv = rsqrtf(smem[0] * (1.f / DDIM) + 1e-5f);
  __syncthreads();

  float part[16];
#pragma unroll
  for (int q = 0; q < 16; ++q) part[q] = 0.f;
#pragma unroll
  for (int i = 0; i < 8; ++i) {
    const int j = tid + i * 256;
    const float x = v[i] * inv * norm_w[j];
    xq[(size_t)row * DDIM + j] = (__bf16)x;
#pragma unroll
    for (int q = 0; q < 16; ++q)
      part[q] += x * gk_w1[q * DDIM + j];
  }
#pragma unroll
  for (int q = 0; q < 16; ++q) smem[tid * 16 + q] = part[q];
  __syncthreads();
  for (int off = 128; off > 0; off >>= 1) {
    if (tid < off) {
#pragma unroll
      for (int q = 0; q < 16; ++q)
        smem[tid * 16 + q] += smem[(tid + off) * 16 + q];
    }
    __syncthreads();
  }
  float hg[16];
#pragma unroll
  for (int q = 0; q < 16; ++q) hg[q] = smem[q];

#pragma unroll
  for (int cc = 0; cc < 4; ++cc) {
    const int c = tid + cc * 256;
    float s = gk_b[c];
#pragma unroll
    for (int q = 0; q < 16; ++q) s += hg[q] * gk_w2[c * 16 + q];
    const float ls = fminf(s, 0.f) - __logf(1.f + __expf(-fabsf(s)));
    gk[(size_t)row * NK + c] = ls * (1.f / 16.f);
  }
}

// ---------------- K2: cumsum over T, rewrite gk buffer as decay ----------------
__global__ void cumsum_decay(float* __restrict__ gk, float* __restrict__ expg) {
  const int idx = blockIdx.x * blockDim.x + threadIdx.x;  // 0..4095
  const int b = idx >> 10, c = idx & (NK - 1);
  float* p = gk + (size_t)b * TDIM * NK + c;
  float tot = 0.f;
  for (int t = 0; t < TDIM; ++t) tot += p[(size_t)t * NK];
  float run = 0.f;
  for (int t = 0; t < TDIM; ++t) {
    run += p[(size_t)t * NK];
    p[(size_t)t * NK] = __expf(tot - run);   // decay in (0,1]
  }
  expg[(size_t)b * NK + c] = __expf(tot);
}

// ---------------- K3: LDS-staged bf16 WMMA GEMM ----------------
// outT[n][row] = (x @ W^T)[row][n] * scale
// block: 256 thr = 8 waves; block tile 128x128; K staged 64 at a time via
// async global->LDS copies, double buffered. Wave computes 32 rows x 64 cols.
#define KSTEP 64
#define LSTRIDE 72   // bf16 elems per LDS row (64 + 8 pad) = 144 B, 16B-aligned

__global__ void __launch_bounds__(256) gemm_xw_bf16(
    const __bf16* __restrict__ X,        // [BT][D]
    const __bf16* __restrict__ W,        // [N][D]
    const float*  __restrict__ dscale,   // [BT][NK] decay, or nullptr
    __bf16* __restrict__ outT)           // [N][BT]
{
  __shared__ __align__(16) __bf16 Xs[2][128 * LSTRIDE];  // 2 x 18 KiB
  __shared__ __align__(16) __bf16 Ws[2][128 * LSTRIDE];  // 2 x 18 KiB

  const int tid = threadIdx.x;
  const int wv = tid >> 5, ln = tid & 31;
  const int lh = ln >> 4, ll = ln & 15;
  const int rowBlk = blockIdx.x * 128;
  const int colBlk = blockIdx.y * 128;
  const int waveM = (wv >> 1) * 32;
  const int waveN = (wv & 1) * 64;

  // stage one 128x64 tile of X and W into LDS buffer `buf` (8 asyncs/thread)
  auto stage = [&](int buf, int kk) {
#pragma unroll
    for (int it = 0; it < 4; ++it) {
      const int c = tid + it * 256;            // chunk id 0..1023
      const int r = c >> 3, kc = (c & 7) * 8;  // 8 bf16 per chunk
      async_ld16(X + (size_t)(rowBlk + r) * DDIM + kk + kc,
                 &Xs[buf][r * LSTRIDE + kc]);
    }
#pragma unroll
    for (int it = 0; it < 4; ++it) {
      const int c = tid + it * 256;
      const int r = c >> 3, kc = (c & 7) * 8;
      async_ld16(W + (size_t)(colBlk + r) * DDIM + kk + kc,
                 &Ws[buf][r * LSTRIDE + kc]);
    }
  };

  v8f acc[2][4] = {};
  stage(0, 0);
  int buf = 0;

  for (int kk = 0; kk < DDIM; kk += KSTEP) {
    if (kk + KSTEP < DDIM) {
      stage(buf ^ 1, kk + KSTEP);
      WAIT_ASYNC(8);   // async loads retire in order: current tile is resident
    } else {
      WAIT_ASYNC(0);
    }
    __syncthreads();

#pragma unroll
    for (int ks = 0; ks < KSTEP; ks += 32) {
      Frag a[2];
#pragma unroll
      for (int mi = 0; mi < 2; ++mi) {
        const __bf16* pa = &Xs[buf][(waveM + mi * 16 + ll) * LSTRIDE + ks + lh * 8];
        a[mi].q[0] = *reinterpret_cast<const uint4*>(pa);
        a[mi].q[1] = *reinterpret_cast<const uint4*>(pa + 16);
      }
#pragma unroll
      for (int j = 0; j < 4; ++j) {
        const __bf16* pb = &Ws[buf][(waveN + j * 16 + ll) * LSTRIDE + ks + lh * 8];
        Frag b;
        b.q[0] = *reinterpret_cast<const uint4*>(pb);
        b.q[1] = *reinterpret_cast<const uint4*>(pb + 16);
#pragma unroll
        for (int mi = 0; mi < 2; ++mi)
          acc[mi][j] = __builtin_amdgcn_wmma_f32_16x16x32_bf16(
              false, a[mi].v, false, b.v, (short)0, acc[mi][j], false, false);
      }
    }
    __syncthreads();
    buf ^= 1;
  }

#pragma unroll
  for (int mi = 0; mi < 2; ++mi) {
    const int rstart = rowBlk + waveM + mi * 16 + lh * 8;
#pragma unroll
    for (int j = 0; j < 4; ++j) {
      const int n = colBlk + waveN + j * 16 + ll;
      Pack8 pk;
#pragma unroll
      for (int p = 0; p < 8; ++p) {
        float vv = acc[mi][j][p];
        if (dscale) vv *= dscale[(size_t)(rstart + p) * NK + n];
        pk.h[p] = (__bf16)vv;
      }
      *reinterpret_cast<uint4*>(outT + (size_t)n * BT + rstart) = pk.q;
    }
  }
}

// ---------------- K4: state update  out = exp(g_total)*state + (k*decay)^T @ v ----------------
__global__ void __launch_bounds__(256) state_update(
    const __bf16* __restrict__ kT,    // [NK][BT]  (c*BT + b*T + t)
    const __bf16* __restrict__ vT,    // [NV][BT]
    const float* __restrict__ state,  // [B][H][DK][DV]
    const float* __restrict__ expg,   // [B][NK]
    float* __restrict__ out)          // [B][H][DK][DV]
{
  const int b = blockIdx.x >> 4;
  const int h = blockIdx.x & 15;
  const int tid = threadIdx.x;
  const int wv = tid >> 5, ln = tid & 31;
  const int lh = ln >> 4, ll = ln & 15;
  const int mBase = (wv >> 1) * 16;  // dk tile: 0,16,32,48
  const int nBase = (wv & 1) * 64;   // dv half: 0,64

  v8f acc[4] = {};
  const __bf16* kb = kT + (size_t)(h * DK + mBase + ll) * BT + (size_t)b * TDIM + lh * 8;

  for (int t0 = 0; t0 < TDIM; t0 += 32) {
    Frag a;
    a.q[0] = ld16(kb + t0);
    a.q[1] = ld16(kb + t0 + 16);
#pragma unroll
    for (int j = 0; j < 4; ++j) {
      const __bf16* pb = vT + (size_t)(h * DV + nBase + j * 16 + ll) * BT
                            + (size_t)b * TDIM + t0 + lh * 8;
      Frag bf;
      bf.q[0] = ld16(pb);
      bf.q[1] = ld16(pb + 16);
      acc[j] = __builtin_amdgcn_wmma_f32_16x16x32_bf16(
          false, a.v, false, bf.v, (short)0, acc[j], false, false);
    }
  }

#pragma unroll
  for (int j = 0; j < 4; ++j) {
    const int n = nBase + j * 16 + ll;
#pragma unroll
    for (int p = 0; p < 8; ++p) {
      const int m = mBase + p + lh * 8;
      const size_t oidx = (((size_t)b * HB + h) * DK + m) * DV + n;
      out[oidx] = expg[(size_t)b * NK + h * DK + m] * state[oidx] + acc[j][p];
    }
  }
}

extern "C" void kernel_launch(void* const* d_in, const int* in_sizes, int n_in,
                              void* d_out, int out_size, void* d_ws, size_t ws_size,
                              hipStream_t stream) {
  const float* hs     = (const float*)d_in[0];
  const float* state  = (const float*)d_in[1];
  const float* norm_w = (const float*)d_in[2];
  const float* k_w    = (const float*)d_in[3];
  const float* v_w    = (const float*)d_in[4];
  const float* gk_w1  = (const float*)d_in[5];
  const float* gk_w2  = (const float*)d_in[6];
  const float* gk_b   = (const float*)d_in[7];
  float* out = (float*)d_out;

  char* ws = (char*)d_ws;
  __bf16* xq  = (__bf16*)(ws + OFF_X);
  __bf16* kwb = (__bf16*)(ws + OFF_KW);
  __bf16* vwb = (__bf16*)(ws + OFF_VW);
  float*  gk  = (float*)(ws + OFF_GK);
  __bf16* kdT = (__bf16*)(ws + OFF_KD);
  __bf16* vT  = (__bf16*)(ws + OFF_VT);
  float*  eg  = (float*)(ws + OFF_EG);

  cvt_f32_bf16<<<(NK * DDIM) / 256, 256, 0, stream>>>(k_w, kwb, NK * DDIM);
  cvt_f32_bf16<<<(NV * DDIM) / 256, 256, 0, stream>>>(v_w, vwb, NV * DDIM);

  rmsnorm_gate<<<BT, 256, 0, stream>>>(hs, norm_w, gk_w1, gk_w2, gk_b, xq, gk);

  cumsum_decay<<<16, 256, 0, stream>>>(gk, eg);

  // k-projection fused with decay; transposed bf16 output
  gemm_xw_bf16<<<dim3(BT / 128, NK / 128), 256, 0, stream>>>(xq, kwb, gk, kdT);
  // v-projection; transposed bf16 output
  gemm_xw_bf16<<<dim3(BT / 128, NV / 128), 256, 0, stream>>>(xq, vwb, nullptr, vT);

  state_update<<<BDIM * HB, 256, 0, stream>>>(kdT, vT, state, eg, out);
}
